// AttentionBlock_43361989820692
// MI455X (gfx1250) — matrix operand order repaired and checked by reading, not compiled, not run on hardware
//
#include <hip/hip_runtime.h>
#include <hip/hip_bf16.h>

typedef __attribute__((ext_vector_type(16))) _Float16 v16h;
typedef __attribute__((ext_vector_type(8)))  _Float16 v8h;
typedef __attribute__((ext_vector_type(8)))  float    v8f;
typedef int v4i __attribute__((vector_size(16)));

#define WMMA_F16(a, b, c) \
    __builtin_amdgcn_wmma_f32_16x16x32_f16(false, (a), false, (b), (short)0, (c), false, false)

#ifndef __has_builtin
#define __has_builtin(x) 0
#endif
#if defined(__HIP_DEVICE_COMPILE__) && __has_builtin(__builtin_amdgcn_global_load_async_to_lds_b128)
#define USE_ASYNC_LDS 1
#else
#define USE_ASYNC_LDS 0
#endif

#if USE_ASYNC_LDS
// async Global -> LDS copy of 16 bytes (bypasses VGPRs, tracked by ASYNCcnt)
__device__ __forceinline__ void async_cp16(_Float16* lds, const _Float16* g) {
    __builtin_amdgcn_global_load_async_to_lds_b128(
        (__attribute__((address_space(1))) v4i*)g,
        (__attribute__((address_space(3))) v4i*)lds, 0, 0);
}
__device__ __forceinline__ void async_wait0() {
#if __has_builtin(__builtin_amdgcn_s_wait_asynccnt)
    __builtin_amdgcn_s_wait_asynccnt(0);
#else
    asm volatile("s_wait_asynccnt 0" ::: "memory");
#endif
}
#endif

// A-fragment (16x32, M x K): lane m = L%16, kh = L/16.
// elems 0..7  <-> K = kc + 8*kh + 0..7 ; elems 8..15 <-> K = kc + 16 + 8*kh + 0..7
__device__ __forceinline__ v16h ld_frag_a(const _Float16* row, int kc, int kh) {
    v8h lo = *(const v8h*)(row + kc + 8 * kh);
    v8h hi = *(const v8h*)(row + kc + 16 + 8 * kh);
    return __builtin_shufflevector(lo, hi, 0,1,2,3,4,5,6,7,8,9,10,11,12,13,14,15);
}
// B-fragment (32x16, K x N): lane n = L%16, kh = L/16; elems 0..7 <-> K = kc + 8*kh + 0..7 ; elems 8..15 <-> K = kc + 16 + 8*kh + 0..7
__device__ __forceinline__ v16h ld_frag_b(const _Float16* row, int kc, int kh) {
    v8h lo = *(const v8h*)(row + kc + 8 * kh);
    v8h hi = *(const v8h*)(row + kc + 16 + 8 * kh);
    return __builtin_shufflevector(lo, hi, 0,1,2,3,4,5,6,7,8,9,10,11,12,13,14,15);
}

// ---------------------------------------------------------------- GroupNorm
__global__ void gn_stats_kernel(const float* __restrict__ x,
                                float* __restrict__ meanb, float* __restrict__ rstdb) {
    __shared__ float ss[256], sq[256];
    int bg = blockIdx.x;                               // b*8 + g
    const float4* p = (const float4*)(x + (size_t)bg * 16 * 4096);
    float s = 0.f, q = 0.f;
    for (int i = threadIdx.x; i < 16384; i += 256) {
        float4 v = p[i];
        s += v.x + v.y + v.z + v.w;
        q += v.x * v.x + v.y * v.y + v.z * v.z + v.w * v.w;
    }
    ss[threadIdx.x] = s; sq[threadIdx.x] = q;
    __syncthreads();
    for (int off = 128; off > 0; off >>= 1) {
        if ((int)threadIdx.x < off) {
            ss[threadIdx.x] += ss[threadIdx.x + off];
            sq[threadIdx.x] += sq[threadIdx.x + off];
        }
        __syncthreads();
    }
    if (threadIdx.x == 0) {
        float m   = ss[0] * (1.f / 65536.f);
        float var = sq[0] * (1.f / 65536.f) - m * m;
        meanb[bg] = m;
        rstdb[bg] = rsqrtf(var + 1e-5f);
    }
}

// normalize and transpose to pixel-major hT[B][N][C] f16
__global__ void gn_apply_kernel(const float* __restrict__ x,
                                const float* __restrict__ gw, const float* __restrict__ gb,
                                const float* __restrict__ meanb, const float* __restrict__ rstdb,
                                _Float16* __restrict__ hT) {
    int gi  = blockIdx.x * 256 + threadIdx.x;
    int e   = gi * 4;
    int b   = e >> 19;
    int rem = e & ((1 << 19) - 1);
    int c   = rem >> 12;
    int n   = rem & 4095;
    int bg  = b * 8 + (c >> 4);
    float m = meanb[bg], r = rstdb[bg];
    float w = gw[c] * r;
    float bb = gb[c] - m * w;
    float4 v = *(const float4*)(x + e);
    _Float16* dst = hT + ((size_t)b * 4096 + n) * 128 + c;
    dst[0]   = (_Float16)(v.x * w + bb);
    dst[128] = (_Float16)(v.y * w + bb);
    dst[256] = (_Float16)(v.z * w + bb);
    dst[384] = (_Float16)(v.w * w + bb);
}

// ---------------------------------------------------------------- QKV GEMM
__global__ void qkv_gemm_kernel(const float* __restrict__ W, const float* __restrict__ bias,
                                const _Float16* __restrict__ hT,
                                _Float16* __restrict__ qT, _Float16* __restrict__ kT,
                                _Float16* __restrict__ vCM) {
    __shared__ __align__(16) _Float16 As[64][136];
    __shared__ __align__(16) _Float16 Bs[64][136];
    int b = blockIdx.z, o_base = blockIdx.y * 64, n_base = blockIdx.x * 64;
    int tid = threadIdx.x;
    for (int i = tid; i < 8192; i += 256) {
        int r = i >> 7, c = i & 127;
        As[r][c] = (_Float16)W[(o_base + r) * 128 + c];
    }
    const unsigned int* src = (const unsigned int*)(hT + ((size_t)b * 4096 + n_base) * 128);
    for (int i = tid; i < 4096; i += 256) {
        int r = i >> 6, d = i & 63;
        *(unsigned int*)&Bs[r][2 * d] = src[r * 64 + d];
    }
    __syncthreads();

    int w = tid >> 5, lane = tid & 31;
    int mt = w & 3, nt0 = (w >> 2) * 2, lr = lane & 15, hi = lane >> 4;
    v8f acc[2] = {};
#pragma unroll
    for (int kc = 0; kc < 128; kc += 32) {
        v16h a  = ld_frag_a(&As[mt * 16 + lr][0], kc, hi);
        v16h b0 = ld_frag_b(&Bs[nt0 * 16 + lr][0], kc, hi);
        v16h b1 = ld_frag_b(&Bs[(nt0 + 1) * 16 + lr][0], kc, hi);
        acc[0] = WMMA_F16(a, b0, acc[0]);
        acc[1] = WMMA_F16(a, b1, acc[1]);
    }
    const float scale = 0.08838834764831845f;          // 128^-0.5, folded into q
    // segment (q/k/v) is uniform per block: o_base is 64-aligned, segments 128-wide
    if (o_base < 128) {
#pragma unroll
        for (int t = 0; t < 2; t++)
#pragma unroll
            for (int r = 0; r < 8; r++) {
                int o = o_base + mt * 16 + r + 8 * hi;
                int n = n_base + (nt0 + t) * 16 + lr;
                qT[((size_t)b * 4096 + n) * 128 + o] = (_Float16)((acc[t][r] + bias[o]) * scale);
            }
    } else if (o_base < 256) {
#pragma unroll
        for (int t = 0; t < 2; t++)
#pragma unroll
            for (int r = 0; r < 8; r++) {
                int o = o_base + mt * 16 + r + 8 * hi;
                int n = n_base + (nt0 + t) * 16 + lr;
                kT[((size_t)b * 4096 + n) * 128 + (o - 128)] = (_Float16)(acc[t][r] + bias[o]);
            }
    } else {
#pragma unroll
        for (int t = 0; t < 2; t++)
#pragma unroll
            for (int r = 0; r < 8; r++) {
                int o = o_base + mt * 16 + r + 8 * hi;
                int n = n_base + (nt0 + t) * 16 + lr;
                vCM[((size_t)b * 128 + (o - 256)) * 4096 + n] = (_Float16)(acc[t][r] + bias[o]);
            }
    }
}

// ---------------------------------------------------------------- Flash attention
__global__ void flash_attn_kernel(const _Float16* __restrict__ qT,
                                  const _Float16* __restrict__ kT,
                                  const _Float16* __restrict__ vCM,
                                  _Float16* __restrict__ attnT) {
    __shared__ __align__(16) _Float16 Qs[64][136];     // [q][c]
    __shared__ __align__(16) _Float16 Ks[64][136];     // [key][c]
    __shared__ __align__(16) _Float16 Vs[128][72];     // [c][key]
    __shared__ __align__(16) _Float16 Ps[64][72];      // [q][key]
    __shared__ float mrow[64], lrow[64], arow[64], red[256];

    int b = blockIdx.y, q0 = blockIdx.x * 64, tid = threadIdx.x;
    const unsigned int* qsrc = (const unsigned int*)(qT + ((size_t)b * 4096 + q0) * 128);
    for (int i = tid; i < 4096; i += 256) {
        int r = i >> 6, d = i & 63;
        *(unsigned int*)&Qs[r][2 * d] = qsrc[r * 64 + d];
    }
    if (tid < 64) { mrow[tid] = -3.0e38f; lrow[tid] = 0.f; }

    int w = tid >> 5, lane = tid & 31;
    int mt = w & 3, nt0 = (w >> 2) * 2, cb = (w >> 2) * 64, lr = lane & 15, hi = lane >> 4;
    int srow = tid >> 2, sseg = tid & 3;
    v8f oacc[4] = {};

    for (int j = 0; j < 64; j++) {
        int k0 = j * 64;
        __syncthreads();                               // previous-iter reads done
#if USE_ASYNC_LDS
        // K tile: 64 rows x 128 halves = 1024 x 16B chunks, async direct-to-LDS
        for (int i = tid; i < 1024; i += 256) {
            int r = i >> 4, q = i & 15;
            async_cp16(&Ks[r][q * 8], kT + ((size_t)b * 4096 + k0 + r) * 128 + q * 8);
        }
        // V tile: 128 rows x 64 halves = 1024 x 16B chunks
        for (int i = tid; i < 1024; i += 256) {
            int c = i >> 3, q = i & 7;
            async_cp16(&Vs[c][q * 8], vCM + ((size_t)b * 128 + c) * 4096 + k0 + q * 8);
        }
#else
        const unsigned int* ksrc = (const unsigned int*)(kT + ((size_t)b * 4096 + k0) * 128);
        for (int i = tid; i < 4096; i += 256) {
            int r = i >> 6, d = i & 63;
            *(unsigned int*)&Ks[r][2 * d] = ksrc[r * 64 + d];
        }
        const unsigned int* vsrc = (const unsigned int*)(vCM + (size_t)b * 128 * 4096 + k0);
        for (int i = tid; i < 4096; i += 256) {
            int c = i >> 5, d = i & 31;
            *(unsigned int*)&Vs[c][2 * d] = vsrc[c * 2048 + d];
        }
#endif
        // prefetch next iteration's K/V tiles into cache while this tile computes
        if (j < 63) {
            if (tid < 64)
                __builtin_prefetch(kT + ((size_t)b * 4096 + k0 + 64 + tid) * 128, 0, 1);
            else if (tid >= 128)
                __builtin_prefetch(vCM + ((size_t)b * 128 + (tid - 128)) * 4096 + k0 + 64, 0, 1);
        }
#if USE_ASYNC_LDS
        async_wait0();
#endif
        __syncthreads();

        // S = Q^T K  (two 16x16 tiles per wave, K=128 -> 4 chained WMMAs)
        v8f s[2] = {};
#pragma unroll
        for (int kc = 0; kc < 128; kc += 32) {
            v16h a  = ld_frag_a(&Qs[mt * 16 + lr][0], kc, hi);
            v16h b0 = ld_frag_b(&Ks[nt0 * 16 + lr][0], kc, hi);
            v16h b1 = ld_frag_b(&Ks[(nt0 + 1) * 16 + lr][0], kc, hi);
            s[0] = WMMA_F16(a, b0, s[0]);
            s[1] = WMMA_F16(a, b1, s[1]);
        }
#pragma unroll
        for (int t = 0; t < 2; t++)
#pragma unroll
            for (int r = 0; r < 8; r++)
                Ps[mt * 16 + r + 8 * hi][(nt0 + t) * 16 + lr] = (_Float16)s[t][r];
        __syncthreads();

        // online softmax, phase A: row max (4 threads / row)
        float mx = -3.0e38f;
#pragma unroll
        for (int c2 = 0; c2 < 16; c2++)
            mx = fmaxf(mx, (float)Ps[srow][sseg * 16 + c2]);
        red[tid] = mx;
        __syncthreads();
        if (tid < 64) {
            float m_old = mrow[tid];
            float mn = fmaxf(fmaxf(red[tid * 4], red[tid * 4 + 1]),
                             fmaxf(red[tid * 4 + 2], red[tid * 4 + 3]));
            mn = fmaxf(m_old, mn);
            mrow[tid] = mn;
            arow[tid] = __expf(m_old - mn);
        }
        __syncthreads();

        // phase B: exponentiate in place + partial row sums
        float mn = mrow[srow], sum = 0.f;
#pragma unroll
        for (int c2 = 0; c2 < 16; c2++) {
            float p = __expf((float)Ps[srow][sseg * 16 + c2] - mn);
            Ps[srow][sseg * 16 + c2] = (_Float16)p;
            sum += p;
        }
        red[tid] = sum;
        __syncthreads();
        if (tid < 64)
            lrow[tid] = lrow[tid] * arow[tid] +
                        red[tid * 4] + red[tid * 4 + 1] + red[tid * 4 + 2] + red[tid * 4 + 3];
        __syncthreads();

        // rescale running O, then O += P @ V^T  (4 c-tiles per wave, K=64)
#pragma unroll
        for (int r = 0; r < 8; r++) {
            float al = arow[mt * 16 + r + 8 * hi];
#pragma unroll
            for (int t = 0; t < 4; t++) oacc[t][r] *= al;
        }
#pragma unroll
        for (int kc = 0; kc < 64; kc += 32) {
            v16h a = ld_frag_a(&Ps[mt * 16 + lr][0], kc, hi);
#pragma unroll
            for (int t = 0; t < 4; t++) {
                v16h bb = ld_frag_b(&Vs[cb + t * 16 + lr][0], kc, hi);
                oacc[t] = WMMA_F16(a, bb, oacc[t]);
            }
        }
    }
    __syncthreads();
#pragma unroll
    for (int r = 0; r < 8; r++) {
        float inv = 1.f / lrow[mt * 16 + r + 8 * hi];
        size_t row = ((size_t)b * 4096 + q0 + mt * 16 + r + 8 * hi) * 128;
#pragma unroll
        for (int t = 0; t < 4; t++)
            attnT[row + cb + t * 16 + lr] = (_Float16)(oacc[t][r] * inv);
    }
}

// ---------------------------------------------------------------- Proj GEMM + residual
__global__ void proj_gemm_kernel(const float* __restrict__ W, const float* __restrict__ bias,
                                 const _Float16* __restrict__ attnT,
                                 const float* __restrict__ x, float* __restrict__ out) {
    __shared__ __align__(16) _Float16 As[64][136];
    __shared__ __align__(16) _Float16 Bs[64][136];
    int b = blockIdx.z, o_base = blockIdx.y * 64, n_base = blockIdx.x * 64;
    int tid = threadIdx.x;
    for (int i = tid; i < 8192; i += 256) {
        int r = i >> 7, c = i & 127;
        As[r][c] = (_Float16)W[(o_base + r) * 128 + c];
    }
    const unsigned int* src = (const unsigned int*)(attnT + ((size_t)b * 4096 + n_base) * 128);
    for (int i = tid; i < 4096; i += 256) {
        int r = i >> 6, d = i & 63;
        *(unsigned int*)&Bs[r][2 * d] = src[r * 64 + d];
    }
    __syncthreads();

    int w = tid >> 5, lane = tid & 31;
    int mt = w & 3, nt0 = (w >> 2) * 2, lr = lane & 15, hi = lane >> 4;
    v8f acc[2] = {};
#pragma unroll
    for (int kc = 0; kc < 128; kc += 32) {
        v16h a  = ld_frag_a(&As[mt * 16 + lr][0], kc, hi);
        v16h b0 = ld_frag_b(&Bs[nt0 * 16 + lr][0], kc, hi);
        v16h b1 = ld_frag_b(&Bs[(nt0 + 1) * 16 + lr][0], kc, hi);
        acc[0] = WMMA_F16(a, b0, acc[0]);
        acc[1] = WMMA_F16(a, b1, acc[1]);
    }
#pragma unroll
    for (int t = 0; t < 2; t++)
#pragma unroll
        for (int r = 0; r < 8; r++) {
            int o = o_base + mt * 16 + r + 8 * hi;
            int n = n_base + (nt0 + t) * 16 + lr;
            size_t idx = ((size_t)b * 128 + o) * 4096 + n;
            out[idx] = acc[t][r] + bias[o] + x[idx];
        }
}

// ---------------------------------------------------------------- launch
extern "C" void kernel_launch(void* const* d_in, const int* in_sizes, int n_in,
                              void* d_out, int out_size, void* d_ws, size_t ws_size,
                              hipStream_t stream) {
    const float* x      = (const float*)d_in[0];
    const float* gn_w   = (const float*)d_in[1];
    const float* gn_b   = (const float*)d_in[2];
    const float* qkv_w  = (const float*)d_in[3];
    const float* qkv_b  = (const float*)d_in[4];
    const float* proj_w = (const float*)d_in[5];
    const float* proj_b = (const float*)d_in[6];
    float* out = (float*)d_out;

    float* meanb = (float*)d_ws;
    float* rstdb = meanb + 64;
    const size_t SZ = (size_t)8 * 4096 * 128;          // elements per f16 tensor
    _Float16* hT    = (_Float16*)((char*)d_ws + 512);
    _Float16* qT    = hT + SZ;
    _Float16* kT    = qT + SZ;
    _Float16* vCM   = kT + SZ;                         // [B][C][N]
    _Float16* attnT = vCM + SZ;                        // [B][N][C]

    gn_stats_kernel<<<dim3(64), dim3(256), 0, stream>>>(x, meanb, rstdb);
    gn_apply_kernel<<<dim3(4096), dim3(256), 0, stream>>>(x, gn_w, gn_b, meanb, rstdb, hT);
    qkv_gemm_kernel<<<dim3(64, 6, 8), dim3(256), 0, stream>>>(qkv_w, qkv_b, hT, qT, kT, vCM);
    flash_attn_kernel<<<dim3(64, 8), dim3(256), 0, stream>>>(qT, kT, vCM, attnT);
    proj_gemm_kernel<<<dim3(64, 2, 8), dim3(256), 0, stream>>>(proj_w, proj_b, attnT, x, out);
}
